// ScaledDotProductAttention_58944131170856
// MI455X (gfx1250) — matrix operand order repaired and checked by reading, not compiled, not run on hardware
//
#include <hip/hip_runtime.h>

// Scaled dot-product attention (flash-attention single pass), CDNA5 / gfx1250.
// B=2, H=16, S=2048, D=64, fp32 in/out, bool mask (true -> score = -10000).
//
// Per workgroup: 8 waves x 16 query rows = 128 rows of one (b,h).
// Inner loop: 32 keys/iter, bf16 WMMA 16x16x32 for Q.K^T and P.V, fp32 accum,
// online softmax in the exp2 domain (log2e folded into the Q scale).
// Row-sums of P are computed by an extra WMMA against a ones-column B tile.
// Row-max uses a v_permlane16_b32 + v_max_num_f32 butterfly in inline asm
// (no tied-operand movs, no re-canonicalization).
// Double-buffered LDS staging (K row-major bf16, V transposed bf16, mask raw
// bytes via global_load_async_to_lds_b128), one barrier per iteration.

#define S_LEN 2048
#define DK    64
#define BM    128
#define BN    32

typedef __attribute__((ext_vector_type(16))) __bf16 bf16x16;
typedef __attribute__((ext_vector_type(8)))  float  f32x8;

union BFrag {
  unsigned int u[8];
  bf16x16 v;
};

__device__ __forceinline__ unsigned short f2bf(float x) {
  union { __bf16 h; unsigned short u; } cv;
  cv.h = (__bf16)x;
  return cv.u;
}
__device__ __forceinline__ unsigned int pk2(float a, float b) {
  return (unsigned int)f2bf(a) | ((unsigned int)f2bf(b) << 16);
}

// single v_max_num_f32 without canonicalization ops
__device__ __forceinline__ float fmax_raw(float a, float b) {
  float d;
  asm("v_max_num_f32 %0, %1, %2" : "=v"(d) : "v"(a), "v"(b));
  return d;
}

// one xor-butterfly step: t = perm16(x); t = max(t, x)   (2 instructions)
template <unsigned LO, unsigned HI>
__device__ __forceinline__ float max_bfly(float x) {
  float t;
  asm("v_permlane16_b32 %0, %1, %2, %3\n\t"
      "v_max_num_f32 %0, %0, %1"
      : "=&v"(t) : "v"(x), "s"(LO), "n"(HI));
  return t;
}
// selectors: 4-bit entries, entry i = i^m, for m = 1,2,4,8
__device__ __forceinline__ float red_max16(float x) {
  x = max_bfly<0x67452301u, 0xEFCDAB89u>(x);
  x = max_bfly<0x54761032u, 0xDCFE98BAu>(x);
  x = max_bfly<0x32107654u, 0xBA98FEDCu>(x);
  x = max_bfly<0xFEDCBA98u, 0x76543210u>(x);
  return x;
}

// v_permlane16_b32 broadcast (lane 0 of each 16-lane row) for the epilogue
__device__ __forceinline__ float bcast16_lane0(float x) {
  int xi = __builtin_bit_cast(int, x);
  int r  = __builtin_amdgcn_permlane16(xi, xi, 0, 0, false, false);
  return __builtin_bit_cast(float, r);
}

// LDS byte layout (32 KB total):
//   Kbuf[buf]  : buf*4096          (32x64 bf16, row-major)
//   Vbuf[buf]  : 8192 + buf*4096   (64x32 bf16, transposed)
//   P scratch  : 16384 + wave*1024 (16x32 bf16 per wave)
//   Maskbuf[b] : 24576 + buf*4096  (128x32 u8, row-major)
#define KO16(buf)  ((buf) * 2048)
#define VO16(buf)  (4096 + (buf) * 2048)
#define PO16(w)    (8192 + (w) * 512)
#define MB8(buf)   (24576 + (buf) * 4096)

#define LOG2E      1.44269504088896340736f
#define QSCALE     (0.125f * LOG2E)          // 1/sqrt(64) * log2(e)
#define MASKVAL    (-14426.950408889634f)    // -10000 * log2(e)

__global__ __launch_bounds__(256)
void flash_attn_wmma_bf16(const float* __restrict__ Q,
                          const float* __restrict__ K,
                          const float* __restrict__ V,
                          const unsigned char* __restrict__ mask,
                          float* __restrict__ Out,
                          int Hn) {
  __shared__ __align__(16) unsigned char smem_raw[32768];
  unsigned short* smem16 = (unsigned short*)smem_raw;
  unsigned int*   smem32 = (unsigned int*)smem_raw;
  unsigned char*  smem8  = (unsigned char*)smem_raw;

  const int tid  = threadIdx.x;
  const int wave = tid >> 5;
  const int lane = tid & 31;
  const int n    = lane & 15;   // A-row / B-col / C-col index
  const int hf   = lane >> 4;   // lane half

  const int bh   = blockIdx.y;
  const int b    = bh / Hn;
  const int qblk = blockIdx.x * BM;

  const size_t bh_base = (size_t)bh * S_LEN * DK;
  const float* Qb = Q + bh_base;
  const float* Kb = K + bh_base;
  const float* Vb = V + bh_base;
  float*       Ob = Out + bh_base;
  const unsigned char* Mb = mask + (size_t)b * S_LEN * S_LEN;

  // each thread stages 16 mask bytes: row = tid>>1, 16-byte half = tid&1
  const int mrow_st = tid >> 1;
  const int mhalf   = tid & 1;
  const unsigned char* mg_base = Mb + (size_t)(qblk + mrow_st) * S_LEN + mhalf * 16;
  const unsigned mlds_row = (unsigned)(mrow_st * 32 + mhalf * 16);

  // ---- async-stage mask block 0 (raw bytes, CDNA5 async-to-LDS path) ----
  {
    unsigned ldsa = (unsigned)(size_t)(void*)&smem8[MB8(0) + mlds_row];
    asm volatile("global_load_async_to_lds_b128 %0, %1, off"
                 :: "v"(ldsa), "v"(mg_base) : "memory");
  }

  // ---- issue K/V block-0 global loads (float4 x2 each) ----
  float4 kr[2], vr[2];
  #pragma unroll
  for (int q = 0; q < 2; ++q) {
    int i4 = tid + q * 256;
    kr[q] = ((const float4*)Kb)[i4];
    vr[q] = ((const float4*)Vb)[i4];
  }

  // ---- phase 1: stage scaled Q tile as bf16 (reuses K/V buffer space) ----
  #pragma unroll
  for (int k = 0; k < (BM * DK / 4) / 256; ++k) {   // 8 float4 per thread
    int i4 = tid + k * 256;
    float4 q4 = ((const float4*)(Qb + (size_t)qblk * DK))[i4];
    smem32[i4 * 2 + 0] = pk2(q4.x * QSCALE, q4.y * QSCALE);
    smem32[i4 * 2 + 1] = pk2(q4.z * QSCALE, q4.w * QSCALE);
  }
  __syncthreads();

  BFrag aQ[2];  // 16x32 A fragments for d=0..31 and d=32..63
  #pragma unroll
  for (int t = 0; t < 2; ++t)
    #pragma unroll
    for (int r = 0; r < 8; ++r) {
      int kk = 16 * hf + 2 * r;  // contiguous k order (k = 16h + s), matches B
      int off16 = (wave * 16 + n) * DK + 32 * t + kk;
      aQ[t].u[r] = smem32[off16 >> 1];
    }
  __syncthreads();

  // ---- store K/V block 0 into buffer 0 ----
  #pragma unroll
  for (int q = 0; q < 2; ++q) {
    int i4  = tid + q * 256;
    int row = (i4 * 4) >> 6;
    int col = (i4 * 4) & 63;
    smem32[(KO16(0) >> 1) + i4 * 2 + 0] = pk2(kr[q].x, kr[q].y);
    smem32[(KO16(0) >> 1) + i4 * 2 + 1] = pk2(kr[q].z, kr[q].w);
    smem16[VO16(0) + (col + 0) * BN + row] = f2bf(vr[q].x);
    smem16[VO16(0) + (col + 1) * BN + row] = f2bf(vr[q].y);
    smem16[VO16(0) + (col + 2) * BN + row] = f2bf(vr[q].z);
    smem16[VO16(0) + (col + 3) * BN + row] = f2bf(vr[q].w);
  }
  asm volatile("s_wait_asynccnt 0x0" ::: "memory");
  __syncthreads();

  // ---- state: row max (log2 domain) + 4 O tiles + 1 row-sum tile ----
  float row_m[8];
  f32x8 oacc[5] = {};              // [0..3] = O 16x64, [4] = l tile (col 0)
  #pragma unroll
  for (int i = 0; i < 8; ++i) row_m[i] = -INFINITY;

  // constant B fragment: column 0 all ones -> P x ones = row sums
  BFrag bOne;
  {
    unsigned ones = (n == 0) ? 0x3F803F80u : 0u;   // two bf16(1.0)
    #pragma unroll
    for (int r = 0; r < 8; ++r) bOne.u[r] = ones;
  }

  const int NBLK = S_LEN / BN;   // 64
  for (int jb = 0; jb < NBLK; ++jb) {
    const int cur = jb & 1;
    const int nxt = cur ^ 1;
    const int j   = jb * BN;
    const bool have_next = (jb + 1) < NBLK;

    // ---- issue next block's loads early (hidden behind WMMA + softmax) ----
    if (have_next) {
      unsigned ldsa = (unsigned)(size_t)(void*)&smem8[MB8(nxt) + mlds_row];
      const unsigned char* mg = mg_base + (j + BN);
      asm volatile("global_load_async_to_lds_b128 %0, %1, off"
                   :: "v"(ldsa), "v"(mg) : "memory");
      #pragma unroll
      for (int q = 0; q < 2; ++q) {
        int i4 = tid + q * 256;
        kr[q] = ((const float4*)(Kb + (size_t)(j + BN) * DK))[i4];
        vr[q] = ((const float4*)(Vb + (size_t)(j + BN) * DK))[i4];
      }
    }

    // ---- scores: c0 = keys j..j+15, c1 = keys j+16..j+31 (Q K^T) ----
    f32x8 c0 = {}, c1 = {};
    #pragma unroll
    for (int t = 0; t < 2; ++t) {
      BFrag b0, b1;
      #pragma unroll
      for (int r = 0; r < 8; ++r) {
        int d = 32 * t + 16 * hf + 2 * r;               // contiguous k order (k = 16h + s), matches A
        b0.u[r] = smem32[(KO16(cur) + (n + 0)  * DK + d) >> 1];
        b1.u[r] = smem32[(KO16(cur) + (n + 16) * DK + d) >> 1];
      }
      c0 = __builtin_amdgcn_wmma_f32_16x16x32_bf16(false, aQ[t].v, false, b0.v,
                                                   (short)0, c0, false, false);
      c1 = __builtin_amdgcn_wmma_f32_16x16x32_bf16(false, aQ[t].v, false, b1.v,
                                                   (short)0, c1, false, false);
    }

    // ---- masked online softmax, exp2 domain (max via asm butterfly) ----
    const int mbase = MB8(cur) + (wave * 16 + 8 * hf) * 32;
    #pragma unroll
    for (int i = 0; i < 8; ++i) {
      float s0 = smem8[mbase + i * 32 + n]      ? MASKVAL : c0[i];
      float s1 = smem8[mbase + i * 32 + 16 + n] ? MASKVAL : c1[i];
      float mx = red_max16(fmax_raw(s0, s1));
      float mn = fmax_raw(row_m[i], mx);
      float sc = __builtin_amdgcn_exp2f(row_m[i] - mn);
      float p0 = __builtin_amdgcn_exp2f(s0 - mn);
      float p1 = __builtin_amdgcn_exp2f(s1 - mn);
      row_m[i] = mn;
      oacc[0][i] *= sc; oacc[1][i] *= sc; oacc[2][i] *= sc;
      oacc[3][i] *= sc; oacc[4][i] *= sc;
      smem16[PO16(wave) + (i + 8 * hf) * BN + n]      = f2bf(p0);
      smem16[PO16(wave) + (i + 8 * hf) * BN + 16 + n] = f2bf(p1);
    }

    // ---- re-gather P as A fragment (same-wave LDS, DS ops in-order) ----
    BFrag aP;
    #pragma unroll
    for (int r = 0; r < 8; ++r) {
      int kk = 16 * hf + 2 * r;
      aP.u[r] = smem32[(PO16(wave) + n * BN + kk) >> 1];
    }

    // ---- O += P * V ; l += P * ones (row sums via the matrix pipe) ----
    #pragma unroll
    for (int d = 0; d < 4; ++d) {
      BFrag bV;
      #pragma unroll
      for (int r = 0; r < 8; ++r) {
        int kk = 16 * hf + 2 * r;
        bV.u[r] = smem32[(VO16(cur) + (d * 16 + n) * BN + kk) >> 1];
      }
      oacc[d] = __builtin_amdgcn_wmma_f32_16x16x32_bf16(false, aP.v, false, bV.v,
                                                        (short)0, oacc[d], false, false);
    }
    oacc[4] = __builtin_amdgcn_wmma_f32_16x16x32_bf16(false, aP.v, false, bOne.v,
                                                      (short)0, oacc[4], false, false);

    // ---- stage next K/V into the other buffer, then single barrier ----
    if (have_next) {
      #pragma unroll
      for (int q = 0; q < 2; ++q) {
        int i4  = tid + q * 256;
        int row = (i4 * 4) >> 6;
        int col = (i4 * 4) & 63;
        smem32[(KO16(nxt) >> 1) + i4 * 2 + 0] = pk2(kr[q].x, kr[q].y);
        smem32[(KO16(nxt) >> 1) + i4 * 2 + 1] = pk2(kr[q].z, kr[q].w);
        smem16[VO16(nxt) + (col + 0) * BN + row] = f2bf(vr[q].x);
        smem16[VO16(nxt) + (col + 1) * BN + row] = f2bf(vr[q].y);
        smem16[VO16(nxt) + (col + 2) * BN + row] = f2bf(vr[q].z);
        smem16[VO16(nxt) + (col + 3) * BN + row] = f2bf(vr[q].w);
      }
      asm volatile("s_wait_asynccnt 0x0" ::: "memory");
    }
    __syncthreads();
  }

  // ---- epilogue: l lives in column 0 of the l-tile; broadcast + divide ----
  #pragma unroll
  for (int i = 0; i < 8; ++i) {
    float l   = bcast16_lane0(oacc[4][i]);
    float inv = 1.0f / l;
    size_t q = (size_t)(qblk + wave * 16 + i + 8 * hf);
    #pragma unroll
    for (int d = 0; d < 4; ++d)
      Ob[q * DK + d * 16 + n] = oacc[d][i] * inv;
  }
}

extern "C" void kernel_launch(void* const* d_in, const int* in_sizes, int n_in,
                              void* d_out, int out_size, void* d_ws, size_t ws_size,
                              hipStream_t stream) {
  const float* Q = (const float*)d_in[0];
  const float* K = (const float*)d_in[1];
  const float* V = (const float*)d_in[2];
  const unsigned char* M = (const unsigned char*)d_in[3];
  float* O = (float*)d_out;

  const int BH = in_sizes[0] / (S_LEN * DK);          // B*H = 32
  const int B  = in_sizes[3] / (S_LEN * S_LEN);       // 2
  const int Hn = BH / B;                              // 16

  dim3 grid(S_LEN / BM, BH);
  flash_attn_wmma_bf16<<<grid, dim3(256), 0, stream>>>(Q, K, V, M, O, Hn);
}
